// RelationFeatureExtractor_44676249813079
// MI455X (gfx1250) — matrix operand order repaired and checked by reading, not compiled, not run on hardware
//
#include <hip/hip_runtime.h>
#include <hip/hip_bf16.h>
#include <math.h>

// ---------------------------------------------------------------------------
// Problem constants
// ---------------------------------------------------------------------------
#define BATCH   512
#define NPAIR   100
#define PAIRS   (BATCH * NPAIR)     // 51200
#define HIDDEN  256
#define NUM_OBJS 81
#define SEM     300
#define K_SUB   320                 // 256 feats + 64 sp          (mult of 32)
#define K_OBJ   416                 // 256 feats + 64 sp + 96 pad (mult of 32)
#define PROB_PAD 96

typedef __attribute__((ext_vector_type(16))) __bf16 v16bf;
typedef __attribute__((ext_vector_type(8)))  float  v8f;
typedef __attribute__((ext_vector_type(4)))  int    v4i;

#define AS1 __attribute__((address_space(1)))
#define AS3 __attribute__((address_space(3)))

// ---- CDNA5 async global->LDS path (builtin confirmed present; v4i* params) ----
#if defined(__has_builtin)
#  if __has_builtin(__builtin_amdgcn_global_load_async_to_lds_b128)
#    define HAVE_ASYNC_LDS 1
#    define ASYNC_CP16(g, l) __builtin_amdgcn_global_load_async_to_lds_b128( \
         (AS1 v4i*)(g), (AS3 v4i*)(l), 0, 0)
#  elif __has_builtin(__builtin_amdgcn_global_async_load_to_lds_b128)
#    define HAVE_ASYNC_LDS 1
#    define ASYNC_CP16(g, l) __builtin_amdgcn_global_async_load_to_lds_b128( \
         (AS1 v4i*)(g), (AS3 v4i*)(l), 0, 0)
#  endif
#  if __has_builtin(__builtin_amdgcn_s_wait_asynccnt)
#    define HAVE_WAIT_ASYNC 1
#  endif
#endif

#ifndef HAVE_ASYNC_LDS
#warning "CDNA5 NOTE: global_load_async_to_lds builtin NOT found -> synchronous LDS staging fallback"
#endif

__device__ __forceinline__ void cp16(const unsigned short* g, unsigned short* l) {
#ifdef HAVE_ASYNC_LDS
    ASYNC_CP16(g, l);
#else
    *(v4i*)l = *(const v4i*)g;
#endif
}

__device__ __forceinline__ void wait_async_future() {   // <=3: one tile in flight
#ifdef HAVE_ASYNC_LDS
#  ifdef HAVE_WAIT_ASYNC
    __builtin_amdgcn_s_wait_asynccnt(3);
#  else
    asm volatile("s_wait_asynccnt 0x3" ::: "memory");
#  endif
#endif
}
__device__ __forceinline__ void wait_async_all() {
#ifdef HAVE_ASYNC_LDS
#  ifdef HAVE_WAIT_ASYNC
    __builtin_amdgcn_s_wait_asynccnt(0);
#  else
    asm volatile("s_wait_asynccnt 0x0" ::: "memory");
#  endif
#endif
}

__device__ __forceinline__ unsigned short f2bf(float f) {
    unsigned u = __float_as_uint(f);
    u = (u + 0x7FFFu + ((u >> 16) & 1u)) >> 16;   // round-to-nearest-even
    return (unsigned short)u;
}

// ---------------------------------------------------------------------------
// Precompute: semW[81x256] = label_emb[81x300] @ obj_w1[320:620, :]   (fp32)
// ---------------------------------------------------------------------------
__global__ void build_semW(const float* __restrict__ label_emb,
                           const float* __restrict__ obj_w1,
                           float* __restrict__ semW) {
    int idx = blockIdx.x * 256 + threadIdx.x;
    if (idx >= NUM_OBJS * 256) return;
    int i = idx >> 8, n = idx & 255;
    float s = 0.f;
    for (int j = 0; j < SEM; ++j)
        s += label_emb[i * SEM + j] * obj_w1[(320 + j) * 256 + n];
    semW[idx] = s;
}

// relT[5x256] = rel_emb[5x300] @ w1_rel_rows + bias  (fp32)
__global__ void build_relT(const float* __restrict__ rel_emb,
                           const float* __restrict__ w1_off,
                           const float* __restrict__ bias,
                           float* __restrict__ relT) {
    int idx = blockIdx.x * 256 + threadIdx.x;
    if (idx >= 5 * 256) return;
    int r = idx >> 8, n = idx & 255;
    float s = bias[n];
    for (int j = 0; j < SEM; ++j) s += rel_emb[r * SEM + j] * w1_off[j * 256 + n];
    relT[idx] = s;
}

// Pack fp32 [rows x 256] block into transposed bf16 [256 x Kpad] at column k0
__global__ void pack_wT(const float* __restrict__ src, unsigned short* __restrict__ dst,
                        int rows, int Kpad, int k0) {
    int idx = blockIdx.x * 256 + threadIdx.x;
    if (idx >= rows * 256) return;
    int r = idx / 256, n = idx % 256;
    dst[(size_t)n * Kpad + k0 + r] = f2bf(src[(size_t)r * 256 + n]);
}

__global__ void zero_wT(unsigned short* __restrict__ dst, int Kpad, int k0, int kcount) {
    int idx = blockIdx.x * 256 + threadIdx.x;
    if (idx >= 256 * kcount) return;
    int n = idx / kcount, k = idx % kcount;
    dst[(size_t)n * Kpad + k0 + k] = 0;
}

// ---------------------------------------------------------------------------
// Featurize: one wave32 per pair.
// ---------------------------------------------------------------------------
__global__ __launch_bounds__(256) void featurize(
    const float* __restrict__ head_boxes, const float* __restrict__ tail_boxes,
    const float* __restrict__ head_feats, const float* __restrict__ tail_feats,
    const float* __restrict__ logits,
    const float* __restrict__ spatial_w, const float* __restrict__ spatial_b,
    unsigned short* __restrict__ Xsub, unsigned short* __restrict__ Xobj,
    int* __restrict__ relIdx)
{
    const int wave = threadIdx.x >> 5;
    const int lane = threadIdx.x & 31;
    const int p = blockIdx.x * 8 + wave;              // 6400*8 == PAIRS exactly

    const float* hb = head_boxes + p * 4;
    const float* tb = tail_boxes + p * 4;
    float hx1 = fminf(fmaxf(hb[0] - 0.5f * hb[2], 0.f), 1.f);
    float hy1 = fminf(fmaxf(hb[1] - 0.5f * hb[3], 0.f), 1.f);
    float hx2 = fminf(fmaxf(hb[0] + 0.5f * hb[2], 0.f), 1.f);
    float hy2 = fminf(fmaxf(hb[1] + 0.5f * hb[3], 0.f), 1.f);
    float tx1 = fminf(fmaxf(tb[0] - 0.5f * tb[2], 0.f), 1.f);
    float ty1 = fminf(fmaxf(tb[1] - 0.5f * tb[3], 0.f), 1.f);
    float tx2 = fminf(fmaxf(tb[0] + 0.5f * tb[2], 0.f), 1.f);
    float ty2 = fminf(fmaxf(tb[1] + 0.5f * tb[3], 0.f), 1.f);

    float dx = 0.5f * (hx1 + hx2) - 0.5f * (tx1 + tx2);
    float dy = 0.5f * (hy1 + hy2) - 0.5f * (ty1 + ty2);
    float dis = sqrtf(dx * dx + dy * dy);
    float theta = atan2f(dy, dx) * 0.31830988618379067f;
    float area_h = (hx2 - hx1) * (hy2 - hy1);
    float area_t = (tx2 - tx1) * (ty2 - ty1);
    float ov = fmaxf(fminf(hx2, tx2) - fmaxf(hx1, tx1), 0.f) *
               fmaxf(fminf(hy2, ty2) - fmaxf(hy1, ty1), 0.f);
    float un = fmaxf(fmaxf(hx2, tx2) - fminf(hx1, tx1), 1.f) *
               fmaxf(fmaxf(hy2, ty2) - fminf(hy1, ty1), 1.f);
    float f8[8] = {dx, dy, dis, theta, ov, un, area_h, area_t};

    bool above  = hy1 > ty1 + 0.5f * ty2;
    bool below  = hy1 < ty1 - 0.5f * ty2;
    bool band   = (ty1 - 0.5f * ty2 < hy1) && (hy1 < ty1 + 0.5f * ty2);
    bool around = band && ((hx1 < tx1 - 0.5f * tx2) || (hx1 > tx1 + 0.5f * tx2));
    bool within = band && (tx1 - 0.5f * tx2 < hx1) && (hx1 < tx1 + 0.5f * tx2);
    int rel = (above ? 1 : 0) + (below ? 2 : 0) + (around ? 3 : 0) + (within ? 4 : 0);
    if (tx2 * ty2 < 1e-5f) rel = 0;
    if (lane == 0) relIdx[p] = rel;

    const float* lg = logits + (size_t)p * NUM_OBJS;
    float v0 = lg[lane];
    float v1 = lg[lane + 32];
    float v2 = (lane + 64 < NUM_OBJS) ? lg[lane + 64] : -3.4e38f;
    float mx = fmaxf(fmaxf(v0, v1), v2);
    for (int s = 16; s > 0; s >>= 1) mx = fmaxf(mx, __shfl_xor(mx, s, 32));
    float e0 = expf(v0 - mx);
    float e1 = expf(v1 - mx);
    float e2 = (lane + 64 < NUM_OBJS) ? expf(v2 - mx) : 0.f;
    float sum = e0 + e1 + e2;
    for (int s = 16; s > 0; s >>= 1) sum += __shfl_xor(sum, s, 32);
    float inv = 1.f / sum;

    unsigned short* xo = Xobj + (size_t)p * K_OBJ;
    unsigned short* xs = Xsub + (size_t)p * K_SUB;
    xo[320 + lane]      = f2bf(e0 * inv);
    xo[320 + lane + 32] = f2bf(e1 * inv);
    xo[320 + lane + 64] = (lane + 64 < NUM_OBJS) ? f2bf(e2 * inv) : (unsigned short)0;

    for (int t = 0; t < 2; ++t) {
        int j = lane + 32 * t;
        float s = spatial_b[j];
        #pragma unroll
        for (int i = 0; i < 8; ++i) s += f8[i] * spatial_w[i * 64 + j];
        unsigned short bv = f2bf(s);
        xs[256 + j] = bv;
        xo[256 + j] = bv;
    }

    // feats copy, widened: float4 in, ushort4 out
    const float4* hf = (const float4*)(head_feats + (size_t)p * HIDDEN);
    const float4* tf = (const float4*)(tail_feats + (size_t)p * HIDDEN);
    #pragma unroll
    for (int t = 0; t < 2; ++t) {
        int i4 = lane + 32 * t;                 // 0..63 float4 chunks
        float4 a = hf[i4], b = tf[i4];
        ushort4 ua = { f2bf(a.x), f2bf(a.y), f2bf(a.z), f2bf(a.w) };
        ushort4 ub = { f2bf(b.x), f2bf(b.y), f2bf(b.z), f2bf(b.w) };
        *(ushort4*)(xs + i4 * 4) = ua;
        *(ushort4*)(xo + i4 * 4) = ub;
    }
}

// ---------------------------------------------------------------------------
// WMMA GEMM: C[P x 256] = X[P x K](bf16) @ W (WT[256 x K] bf16)
// block = 256 threads = 8 waves; tile 64(M) x 128(N); K-step 32;
// double-buffered LDS, async global->LDS prefetch (2-deep pipeline).
// addMode: 0 -> +bias[col] ; 1 -> +relT[relIdx[row]*256+col]
// outMode: 0 -> Hout=bf16(relu) ; 1 -> Fout=0.5*relu ; 2 -> Fout+=0.5*relu
// ---------------------------------------------------------------------------
#define LDSS 40   // 32 + 8 pad -> 80B row stride: 16B-aligned & bank-conflict-free

__device__ __forceinline__ void stage_tile(
    const unsigned short* __restrict__ X, const unsigned short* __restrict__ WT,
    int K, int rowBase, int colBase, int k0,
    unsigned short* bufA, unsigned short* bufB, int tid)
{
    const int r  = tid >> 2;               // 0..63
    const int kc = (tid & 3) << 3;         // 0,8,16,24 (ushorts)
    cp16(X  + (size_t)(rowBase + r) * K + k0 + kc, bufA + r * LDSS + kc);          // A: 1 op/wave
    cp16(WT + (size_t)(colBase + r) * K + k0 + kc, bufB + r * LDSS + kc);          // B lo
    cp16(WT + (size_t)(colBase + 64 + r) * K + k0 + kc, bufB + (64 + r) * LDSS + kc); // B hi
}

__global__ __launch_bounds__(256) void gemm_wmma(
    const unsigned short* __restrict__ X,
    const unsigned short* __restrict__ WT,
    int K,
    const float* __restrict__ relT, const int* __restrict__ relIdx,
    const float* __restrict__ bias,
    int addMode, int outMode,
    unsigned short* __restrict__ Hout, float* __restrict__ Fout)
{
    __shared__ __align__(16) unsigned short ldsA[2][64 * LDSS];
    __shared__ __align__(16) unsigned short ldsB[2][128 * LDSS];

    const int tid  = threadIdx.x;
    const int lane = tid & 31;
    const int wave = tid >> 5;
    const int wm = wave & 3;               // 4 M-subtiles of 16
    const int wn = wave >> 2;              // 2 N-halves of 64 (4 wmma each)
    const int rowBase = blockIdx.y * 64;
    const int colBase = blockIdx.x * 128;

    v8f acc[4] = {};
    const int nk = K >> 5;

    stage_tile(X, WT, K, rowBase, colBase, 0,  ldsA[0], ldsB[0], tid);
    stage_tile(X, WT, K, rowBase, colBase, 32, ldsA[1], ldsB[1], tid);

    const int m    = wm * 16 + (lane & 15);
    const int kb   = (lane >> 4) * 8;      // A: lanes 0-15 K{0-7,16-23}; 16-31 K{8-15,24-31}
    const int koff = (lane >> 4) * 16;     // unused; B reads use kb, same K order as A

    for (int t = 0; t < nk; ++t) {
        if (t + 1 < nk) wait_async_future(); else wait_async_all();
        __syncthreads();

        const unsigned short* bufA = ldsA[t & 1];
        const unsigned short* bufB = ldsB[t & 1];

        union { v16bf v; v4i q[2]; } A;
        const unsigned short* aRow = bufA + m * LDSS;
        A.q[0] = *(const v4i*)(aRow + kb);
        A.q[1] = *(const v4i*)(aRow + 16 + kb);

        #pragma unroll
        for (int j = 0; j < 4; ++j) {
            union { v16bf v; v4i q[2]; } B;
            const unsigned short* bRow = bufB + (wn * 64 + j * 16 + (lane & 15)) * LDSS;
            B.q[0] = *(const v4i*)(bRow + kb);
            B.q[1] = *(const v4i*)(bRow + 16 + kb);
            acc[j] = __builtin_amdgcn_wmma_f32_16x16x32_bf16(false, A.v, false, B.v,
                                                             (short)0, acc[j], false, false);
        }
        __syncthreads();
        if (t + 2 < nk)
            stage_tile(X, WT, K, rowBase, colBase, (t + 2) * 32,
                       (unsigned short*)ldsA[t & 1], (unsigned short*)ldsB[t & 1], tid);
    }

    // epilogue: D layout -> row = base + wm*16 + (lane>>4)*8 + v ; col = lane&15
    const int mrow = rowBase + wm * 16 + (lane >> 4) * 8;
    const int cb = colBase + wn * 64 + (lane & 15);
    #pragma unroll
    for (int v = 0; v < 8; ++v) {
        int row = mrow + v;
        const float* add = (addMode == 1) ? (relT + relIdx[row] * 256) : bias;
        #pragma unroll
        for (int j = 0; j < 4; ++j) {
            int c = cb + j * 16;
            float x = fmaxf(acc[j][v] + add[c], 0.f);
            size_t o = (size_t)row * 256 + c;
            if (outMode == 0)      Hout[o] = f2bf(x);
            else if (outMode == 1) Fout[o] = 0.5f * x;
            else                   Fout[o] += 0.5f * x;
        }
    }
}

// ---------------------------------------------------------------------------
extern "C" void kernel_launch(void* const* d_in, const int* in_sizes, int n_in,
                              void* d_out, int out_size, void* d_ws, size_t ws_size,
                              hipStream_t stream) {
    const float* head_boxes = (const float*)d_in[0];
    const float* tail_boxes = (const float*)d_in[1];
    const float* head_feats = (const float*)d_in[2];
    const float* tail_feats = (const float*)d_in[3];
    const float* logits     = (const float*)d_in[4];
    const float* spatial_w  = (const float*)d_in[5];
    const float* spatial_b  = (const float*)d_in[6];
    const float* label_emb  = (const float*)d_in[7];
    const float* rel_emb    = (const float*)d_in[8];
    const float* sub_w1     = (const float*)d_in[9];
    const float* sub_b1     = (const float*)d_in[10];
    const float* sub_w2     = (const float*)d_in[11];
    const float* sub_b2     = (const float*)d_in[12];
    const float* obj_w1     = (const float*)d_in[13];
    const float* obj_b1     = (const float*)d_in[14];
    const float* obj_w2     = (const float*)d_in[15];
    const float* obj_b2     = (const float*)d_in[16];
    float* out = (float*)d_out;

    char* ws = (char*)d_ws;
    auto alloc = [&](size_t bytes) -> void* {
        void* p = (void*)ws;
        ws += (bytes + 255) & ~(size_t)255;
        return p;
    };
    unsigned short* Xsub  = (unsigned short*)alloc((size_t)PAIRS * K_SUB * 2);
    unsigned short* Xobj  = (unsigned short*)alloc((size_t)PAIRS * K_OBJ * 2);
    unsigned short* H     = (unsigned short*)alloc((size_t)PAIRS * 256 * 2);
    int*            relIdx= (int*)alloc((size_t)PAIRS * 4);
    unsigned short* WTsub = (unsigned short*)alloc((size_t)K_SUB * 256 * 2);
    unsigned short* WTobj = (unsigned short*)alloc((size_t)K_OBJ * 256 * 2);
    unsigned short* WT2s  = (unsigned short*)alloc((size_t)256 * 256 * 2);
    unsigned short* WT2o  = (unsigned short*)alloc((size_t)256 * 256 * 2);
    float*          semW  = (float*)alloc((size_t)NUM_OBJS * 256 * 4);
    float*          relWs = (float*)alloc((size_t)5 * 256 * 4);
    float*          relWo = (float*)alloc((size_t)5 * 256 * 4);

    build_semW<<<(NUM_OBJS * 256 + 255) / 256, 256, 0, stream>>>(label_emb, obj_w1, semW);
    build_relT<<<(5 * 256 + 255) / 256, 256, 0, stream>>>(rel_emb, sub_w1 + 320 * 256, sub_b1, relWs);
    build_relT<<<(5 * 256 + 255) / 256, 256, 0, stream>>>(rel_emb, obj_w1 + 620 * 256, obj_b1, relWo);

    pack_wT<<<(320 * 256 + 255) / 256, 256, 0, stream>>>(sub_w1, WTsub, 320, K_SUB, 0);
    pack_wT<<<(320 * 256 + 255) / 256, 256, 0, stream>>>(obj_w1, WTobj, 320, K_OBJ, 0);
    pack_wT<<<(NUM_OBJS * 256 + 255) / 256, 256, 0, stream>>>(semW, WTobj, NUM_OBJS, K_OBJ, 320);
    zero_wT<<<(256 * (PROB_PAD - NUM_OBJS) + 255) / 256, 256, 0, stream>>>(
        WTobj, K_OBJ, 320 + NUM_OBJS, PROB_PAD - NUM_OBJS);
    pack_wT<<<(256 * 256 + 255) / 256, 256, 0, stream>>>(sub_w2, WT2s, 256, 256, 0);
    pack_wT<<<(256 * 256 + 255) / 256, 256, 0, stream>>>(obj_w2, WT2o, 256, 256, 0);

    featurize<<<PAIRS / 8, 256, 0, stream>>>(head_boxes, tail_boxes, head_feats,
                                             tail_feats, logits, spatial_w, spatial_b,
                                             Xsub, Xobj, relIdx);

    dim3 grid(256 / 128, PAIRS / 64);   // (2, 800)
    gemm_wmma<<<grid, 256, 0, stream>>>(Xsub, WTsub, K_SUB, relWs, relIdx, nullptr,
                                        1, 0, H, nullptr);
    gemm_wmma<<<grid, 256, 0, stream>>>(H, WT2s, 256, nullptr, nullptr, sub_b2,
                                        0, 1, nullptr, out);
    gemm_wmma<<<grid, 256, 0, stream>>>(Xobj, WTobj, K_OBJ, relWo, relIdx, nullptr,
                                        1, 0, H, nullptr);
    gemm_wmma<<<grid, 256, 0, stream>>>(H, WT2o, 256, nullptr, nullptr, obj_b2,
                                        0, 2, nullptr, out);
}
